// DeformConvBlock_19129784336968
// MI455X (gfx1250) — matrix operand order repaired and checked by reading, not compiled, not run on hardware
//
#include <hip/hip_runtime.h>

typedef __attribute__((ext_vector_type(16))) _Float16 v16h;
typedef __attribute__((ext_vector_type(8)))  _Float16 v8h;
typedef __attribute__((ext_vector_type(8)))  float    v8f;
typedef __attribute__((ext_vector_type(4)))  float    v4f;

#define Bn   4
#define Cn   64
#define Hn   256
#define Wn   256
#define OUTn 64
#define HW   (Hn*Wn)        // 65536
#define NPIX (Bn*HW)        // 262144

// workspace layout (bytes)
#define XT_OFF   0ull                    // f16 [B][H][W][C]       33,554,432 B
#define OFFS_OFF 33554432ull             // f32 [NPIX][18]         18,874,368 B
#define WOFF_OFF 52428800ull             // f16 packed 576x32          36,864 B
#define WDEF_OFF 52465664ull             // f16 packed 576x64          73,728 B

__device__ __forceinline__ int iclamp(int v, int lo, int hi) {
    return v < lo ? lo : (v > hi ? hi : v);
}
__device__ __forceinline__ v8h splat8(_Float16 x) {
    return (v8h){x, x, x, x, x, x, x, x};
}

// ---------------------------------------------------------------------------
// NCHW f32 -> NHWC f16 transpose through LDS.
// grid: B*H*(W/64) = 4096 blocks of 256 threads
// ---------------------------------------------------------------------------
__global__ void dcn_transpose_kernel(const float* __restrict__ x,
                                     _Float16* __restrict__ xT) {
    __shared__ float tile[64][65];
    int blk = blockIdx.x;
    int wt = blk & 3;
    int h  = (blk >> 2) & 255;
    int b  = blk >> 10;
    int w0 = wt * 64;
    int t  = threadIdx.x;

    int wl = t & 63, cq = t >> 6;
    #pragma unroll 4
    for (int cc = 0; cc < 16; ++cc) {
        int c = cc * 4 + cq;
        tile[c][wl] = x[(((size_t)b * Cn + c) * Hn + h) * Wn + w0 + wl];
    }
    __syncthreads();
    int cl = t & 63, wq = t >> 6;
    #pragma unroll 4
    for (int ww = 0; ww < 16; ++ww) {
        int w = ww * 4 + wq;
        xT[((size_t)b * HW + h * Wn + w0 + w) * Cn + cl] = (_Float16)tile[cl][w];
    }
}

// ---------------------------------------------------------------------------
// Pack w_off [18,64,3,3] f32 -> f16 B-fragments, K=576 (tap-major, ch-minor),
// N padded 18->32. Fragment (ktg, t): half index = ((ktg*2+t)*32 + lane)*16 + e
// lane holds column n = t*16 + lane%16, K = ktg*32 + (lane>=16?16:0) + e.
// ---------------------------------------------------------------------------
__global__ void dcn_pack_woff_kernel(const float* __restrict__ w,
                                     _Float16* __restrict__ pk) {
    int idx = blockIdx.x * blockDim.x + threadIdx.x;
    if (idx >= 18 * 2 * 32 * 16) return;
    int e    = idx & 15;
    int lane = (idx >> 4) & 31;
    int ft   = idx >> 9;          // ktg*2 + t
    int t2   = ft & 1;
    int ktg  = ft >> 1;
    int n    = t2 * 16 + (lane & 15);
    int k    = ktg * 32 + ((lane >> 4) * 16) + e;
    int tap  = k >> 6, c = k & 63;
    float v  = (n < 18) ? w[((size_t)n * Cn + c) * 9 + tap] : 0.f;
    pk[idx]  = (_Float16)v;
}

// Pack w_def [64,64,3,3] f32 -> f16 B-fragments, N=64 (4 tiles).
__global__ void dcn_pack_wdef_kernel(const float* __restrict__ w,
                                     _Float16* __restrict__ pk) {
    int idx = blockIdx.x * blockDim.x + threadIdx.x;
    if (idx >= 18 * 4 * 32 * 16) return;
    int e    = idx & 15;
    int lane = (idx >> 4) & 31;
    int ft   = idx >> 9;          // ktg*4 + t
    int t4   = ft & 3;
    int ktg  = ft >> 2;
    int n    = t4 * 16 + (lane & 15);
    int k    = ktg * 32 + ((lane >> 4) * 16) + e;
    int tap  = k >> 6, c = k & 63;
    pk[idx]  = (_Float16)w[((size_t)n * Cn + c) * 9 + tap];
}

// ---------------------------------------------------------------------------
// Async-stage packed weights into LDS with the CDNA5 async global->LDS DMA
// path (global_load_async_to_lds_b128, ASYNCcnt), then barrier.
// iters = bytes/4096 (256 threads x 16B per iteration).
// ---------------------------------------------------------------------------
__device__ __forceinline__ void stage_weights_async(_Float16* lds,
                                                    const _Float16* __restrict__ src,
                                                    int iters) {
    // Opaque never-taken store: the real writes happen via the async DMA in
    // the asm below, which GlobalOpt cannot see. Without a visible store the
    // compiler folds all loads of the shared array to undef.
    if (__builtin_amdgcn_readfirstlane(threadIdx.x) == 0x12345u) lds[0] = (_Float16)0;

    // Low 32 bits of the flat address of a __shared__ object are its LDS
    // byte offset within the workgroup allocation (aperture layout).
    unsigned ldsbase = (unsigned)(uintptr_t)lds;
    unsigned base = ldsbase + (unsigned)threadIdx.x * 16u;
    unsigned long long g = (unsigned long long)(uintptr_t)src
                         + (unsigned long long)threadIdx.x * 16ull;
    for (int i = 0; i < iters; ++i) {
        unsigned off = (unsigned)i * 4096u;
        asm volatile("global_load_async_to_lds_b128 %0, %1, off"
                     :: "v"(base + off), "v"(g + off) : "memory");
    }
    asm volatile("s_wait_asynccnt 0x0" ::: "memory");
    __syncthreads();
}

// ---------------------------------------------------------------------------
// Pass 1: offset conv as shifted-window implicit GEMM (f16 WMMA).
// One wave per 16-pixel tile. 9 taps x 2 K-tiles x 2 N-tiles.
// B-fragments staged in LDS. Output: offs[p][18] f32, pixel-major.
// ---------------------------------------------------------------------------
__global__ void dcn_offset_gemm_kernel(const _Float16* __restrict__ xT,
                                       const _Float16* __restrict__ wpk,
                                       const float* __restrict__ b_off,
                                       float* __restrict__ offs) {
    __shared__ _Float16 ldsW[18 * 2 * 32 * 16];   // 36,864 B
    stage_weights_async(ldsW, wpk, 9);

    int wave = threadIdx.x >> 5;
    int lane = threadIdx.x & 31;
    int tile = blockIdx.x * (blockDim.x >> 5) + wave;
    int wt = tile & 15;
    int h  = (tile >> 4) & 255;
    int b  = tile >> 12;
    int w0 = wt * 16;
    int m  = lane & 15;        // A row (pixel in tile)
    int hi = lane >> 4;        // K-chunk select
    int xw = w0 + m;

    v8f acc0 = {}, acc1 = {};
    const v8h zero8 = {};

    for (int tap = 0; tap < 9; ++tap) {
        int ki = tap / 3, kj = tap % 3;
        int ys = h + ki - 1;
        int xs = xw + kj - 1;
        bool valid = (ys >= 0) & (ys < Hn) & (xs >= 0) & (xs < Wn);
        int ysc = iclamp(ys, 0, Hn - 1);
        int xsc = iclamp(xs, 0, Wn - 1);
        const _Float16* base = xT + ((size_t)b * HW + ysc * Wn + xsc) * Cn;
        #pragma unroll
        for (int kt = 0; kt < 2; ++kt) {
            int c0 = kt * 32 + hi * 16;
            v8h lo = *(const v8h*)(base + c0);
            v8h up = *(const v8h*)(base + c0 + 8);
            lo = valid ? lo : zero8;
            up = valid ? up : zero8;
            v16h a;
            #pragma unroll
            for (int e = 0; e < 8; ++e) { a[e] = lo[e]; a[e + 8] = up[e]; }
            int ktg = tap * 2 + kt;
            v16h bf0 = *(const v16h*)(ldsW + ((ktg * 2 + 0) * 32 + lane) * 16);
            v16h bf1 = *(const v16h*)(ldsW + ((ktg * 2 + 1) * 32 + lane) * 16);
            acc0 = __builtin_amdgcn_wmma_f32_16x16x32_f16(false, a, false, bf0,
                                                          (short)0, acc0, false, false);
            acc1 = __builtin_amdgcn_wmma_f32_16x16x32_f16(false, a, false, bf1,
                                                          (short)0, acc1, false, false);
        }
    }

    int n0 = lane & 15;
    float bias0 = b_off[n0];
    float bias1 = (n0 + 16 < 18) ? b_off[n0 + 16] : 0.f;
    #pragma unroll
    for (int r = 0; r < 8; ++r) {
        int M = r + hi * 8;
        size_t p = (size_t)b * HW + h * Wn + w0 + M;
        offs[p * 18 + n0] = acc0[r] + bias0;
        if (n0 + 16 < 18) offs[p * 18 + 16 + n0] = acc1[r] + bias1;
    }
}

// ---------------------------------------------------------------------------
// Pass 2: deformable gather + main GEMM (f16 WMMA, N=64).
// Bilinear blend in packed half math (v_pk_fma_f16). Arithmetic predication
// only (EXEC stays all-ones for WMMA). Stores as float4 (consecutive w).
// ---------------------------------------------------------------------------
__global__ void dcn_deform_gemm_kernel(const _Float16* __restrict__ xT,
                                       const float* __restrict__ offs,
                                       const _Float16* __restrict__ wpk,
                                       const float* __restrict__ b_def,
                                       float* __restrict__ out) {
    __shared__ _Float16 ldsW[18 * 4 * 32 * 16];   // 73,728 B
    stage_weights_async(ldsW, wpk, 18);

    int wave = threadIdx.x >> 5;
    int lane = threadIdx.x & 31;
    int tile = blockIdx.x * (blockDim.x >> 5) + wave;
    int wt = tile & 15;
    int h  = (tile >> 4) & 255;
    int b  = tile >> 12;
    int w0 = wt * 16;
    int m  = lane & 15;
    int hi = lane >> 4;
    int xw = w0 + m;
    size_t p = (size_t)b * HW + h * Wn + xw;

    v8f acc[4];
    #pragma unroll
    for (int t = 0; t < 4; ++t) acc[t] = (v8f){};

    const _Float16* xb = xT + (size_t)b * HW * Cn;

    for (int tap = 0; tap < 9; ++tap) {
        int ki = tap / 3, kj = tap % 3;
        float oy = offs[p * 18 + 2 * tap];
        float ox = offs[p * 18 + 2 * tap + 1];
        float ys = (float)(h + ki - 1) + oy;
        float xs = (float)(xw + kj - 1) + ox;
        float y0f = floorf(ys), x0f = floorf(xs);
        float ly = ys - y0f, lx = xs - x0f;
        int y0 = (int)y0f, x0 = (int)x0f;
        int y1 = y0 + 1,  x1 = x0 + 1;
        float vy0 = ((y0 >= 0) & (y0 < Hn)) ? 1.f : 0.f;
        float vy1 = ((y1 >= 0) & (y1 < Hn)) ? 1.f : 0.f;
        float vx0 = ((x0 >= 0) & (x0 < Wn)) ? 1.f : 0.f;
        float vx1 = ((x1 >= 0) & (x1 < Wn)) ? 1.f : 0.f;
        // corner weights, converted once per tap to half for packed blending
        v8h W00 = splat8((_Float16)((1.f - ly) * (1.f - lx) * vy0 * vx0));
        v8h W01 = splat8((_Float16)((1.f - ly) * lx         * vy0 * vx1));
        v8h W10 = splat8((_Float16)(ly         * (1.f - lx) * vy1 * vx0));
        v8h W11 = splat8((_Float16)(ly * lx                 * vy1 * vx1));
        int y0c = iclamp(y0, 0, Hn - 1), y1c = iclamp(y1, 0, Hn - 1);
        int x0c = iclamp(x0, 0, Wn - 1), x1c = iclamp(x1, 0, Wn - 1);
        const _Float16* p00 = xb + ((size_t)y0c * Wn + x0c) * Cn;
        const _Float16* p01 = xb + ((size_t)y0c * Wn + x1c) * Cn;
        const _Float16* p10 = xb + ((size_t)y1c * Wn + x0c) * Cn;
        const _Float16* p11 = xb + ((size_t)y1c * Wn + x1c) * Cn;

        #pragma unroll
        for (int kt = 0; kt < 2; ++kt) {
            v16h a;
            #pragma unroll
            for (int ch = 0; ch < 2; ++ch) {
                int c0 = kt * 32 + hi * 16 + ch * 8;
                v8h c00 = *(const v8h*)(p00 + c0);
                v8h c01 = *(const v8h*)(p01 + c0);
                v8h c10 = *(const v8h*)(p10 + c0);
                v8h c11 = *(const v8h*)(p11 + c0);
                // packed-half bilinear blend -> v_pk_fma_f16 chain
                v8h s = c00 * W00 + c01 * W01 + c10 * W10 + c11 * W11;
                #pragma unroll
                for (int e = 0; e < 8; ++e) a[ch * 8 + e] = s[e];
            }
            int ktg = tap * 2 + kt;
            #pragma unroll
            for (int t = 0; t < 4; ++t) {
                v16h bf = *(const v16h*)(ldsW + ((ktg * 4 + t) * 32 + lane) * 16);
                acc[t] = __builtin_amdgcn_wmma_f32_16x16x32_f16(false, a, false, bf,
                                                                (short)0, acc[t], false, false);
            }
        }
    }

    // lane holds n = t*16 + lane%16, w = w0 + hi*8 + r  (8 consecutive w)
    int n0 = lane & 15;
    #pragma unroll
    for (int t = 0; t < 4; ++t) {
        int n = t * 16 + n0;
        float bias = b_def[n];
        float* dst = out + (((size_t)b * OUTn + n) * Hn + h) * Wn + w0 + hi * 8;
        v4f s0 = {acc[t][0] + bias, acc[t][1] + bias, acc[t][2] + bias, acc[t][3] + bias};
        v4f s1 = {acc[t][4] + bias, acc[t][5] + bias, acc[t][6] + bias, acc[t][7] + bias};
        *(v4f*)(dst)     = s0;
        *(v4f*)(dst + 4) = s1;
    }
}

// ---------------------------------------------------------------------------
extern "C" void kernel_launch(void* const* d_in, const int* in_sizes, int n_in,
                              void* d_out, int out_size, void* d_ws, size_t ws_size,
                              hipStream_t stream) {
    const float* x     = (const float*)d_in[0];
    const float* w_off = (const float*)d_in[1];
    const float* b_off = (const float*)d_in[2];
    const float* w_def = (const float*)d_in[3];
    const float* b_def = (const float*)d_in[4];
    float* out = (float*)d_out;

    char* ws = (char*)d_ws;
    _Float16* xT     = (_Float16*)(ws + XT_OFF);
    float*    offs   = (float*)   (ws + OFFS_OFF);
    _Float16* woffpk = (_Float16*)(ws + WOFF_OFF);
    _Float16* wdefpk = (_Float16*)(ws + WDEF_OFF);

    // 1) NCHW f32 -> NHWC f16
    dcn_transpose_kernel<<<Bn * Hn * (Wn / 64), 256, 0, stream>>>(x, xT);
    // 2) weight packing into WMMA B-fragment order
    dcn_pack_woff_kernel<<<(18 * 2 * 32 * 16 + 255) / 256, 256, 0, stream>>>(w_off, woffpk);
    dcn_pack_wdef_kernel<<<(18 * 4 * 32 * 16 + 255) / 256, 256, 0, stream>>>(w_def, wdefpk);
    // 3) offset conv (WMMA implicit GEMM), offsets stored pixel-major
    int tiles = NPIX / 16;                 // 16384 tiles, 8 waves / block
    dcn_offset_gemm_kernel<<<tiles / 8, 256, 0, stream>>>(xT, woffpk, b_off, offs);
    // 4) deformable gather + main WMMA GEMM
    dcn_deform_gemm_kernel<<<tiles / 8, 256, 0, stream>>>(xT, offs, wdefpk, b_def, out);
}
